// TransMIL_11063835754988
// MI455X (gfx1250) — matrix operand order repaired and checked by reading, not compiled, not run on hardware
//
#include <hip/hip_runtime.h>
#include <hip/hip_bf16.h>
#include <stdint.h>

typedef __attribute__((ext_vector_type(16))) _Float16 v16h;
typedef __attribute__((ext_vector_type(8)))  _Float16 v8h;
typedef __attribute__((ext_vector_type(8)))  float    v8f;

// ---- problem constants (shapes fixed by setup_inputs) ----
constexpr int D    = 512;     // model dim
constexpr int HH   = 8;       // heads
constexpr int DH   = 64;      // head dim
constexpr int MM   = 256;     // landmarks
constexpr int NP   = 22528;   // padded sequence (88*256)
constexpr int N1   = 22501;   // tokens incl cls (150*150+1)
constexpr int PADR = 27;      // NP - N1
constexpr int LG   = 88;      // NP / MM
constexpr int HS   = 150;     // spatial side

// =====================================================================
// WMMA 16x16x32 f16 tile GEMM core: one wave computes a 16x64 tile.
// A: row-major [M x K] f16.  B: NT -> Bt row-major [N x K]; NN -> row-major [K x N].
// A/B fragments use contiguous k order, k = 16*(lane/16) + s, in both operands (wave32).
// =====================================================================
template<bool NT>
__device__ inline void gemm_tile(const _Float16* __restrict__ A, int lda,
                                 const _Float16* __restrict__ B, int ldb,
                                 int K, int m0, int n0, v8f acc[4]) {
  const int lane  = threadIdx.x & 31;
  const int arow  = m0 + (lane & 15);
  const int asel  = (lane >> 4) << 4;   // 0 or 16
  const int bcol0 = lane & 15;
  const int bksel = (lane >> 4) << 4;   // 0 or 16
  for (int k0 = 0; k0 < K; k0 += 32) {
    v16h a;
    {
      const _Float16* pa = A + (size_t)arow * lda + k0 + asel;
      v8h lo = *(const v8h*)pa;
      v8h hi = *(const v8h*)(pa + 8);
#pragma unroll
      for (int j = 0; j < 8; ++j) { a[j] = lo[j]; a[j + 8] = hi[j]; }
    }
#pragma unroll
    for (int nt = 0; nt < 4; ++nt) {
      const int ncol = n0 + nt * 16 + bcol0;
      v16h b;
      if (NT) {
        const _Float16* pb = B + (size_t)ncol * ldb + k0 + bksel;
        v8h lo = *(const v8h*)pb;
        v8h hi = *(const v8h*)(pb + 8);
#pragma unroll
        for (int j = 0; j < 8; ++j) { b[j] = lo[j]; b[j + 8] = hi[j]; }
      } else {
        const _Float16* pb = B + (size_t)(k0 + bksel) * ldb + ncol;
#pragma unroll
        for (int j = 0; j < 16; ++j) b[j] = pb[(size_t)j * ldb];
      }
      acc[nt] = __builtin_amdgcn_wmma_f32_16x16x32_f16(
          false, a, false, b, (short)0, acc[nt], false, false);
    }
  }
}

__device__ inline void zero_acc(v8f acc[4]) {
  v8f z = {0.f,0.f,0.f,0.f,0.f,0.f,0.f,0.f};
#pragma unroll
  for (int i = 0; i < 4; ++i) acc[i] = z;
}

// ---- generic gemm, f32 store, batched over blockIdx.z ----
template<bool NT>
__global__ __launch_bounds__(128)
void k_gemm_f32(const _Float16* A, int lda, long long sA,
                const _Float16* B, int ldb, long long sB,
                float* C, int ldc, long long sC, int K, float alpha) {
  const _Float16* Ab = A + (size_t)blockIdx.z * sA;
  const _Float16* Bb = B + (size_t)blockIdx.z * sB;
  float* Cb = C + (size_t)blockIdx.z * sC;
  const int wave = threadIdx.x >> 5;
  const int m0 = (blockIdx.y * 4 + wave) * 16;
  const int n0 = blockIdx.x * 64;
  v8f acc[4]; zero_acc(acc);
  gemm_tile<NT>(Ab, lda, Bb, ldb, K, m0, n0, acc);
  const int lane = threadIdx.x & 31;
  const int mb = m0 + ((lane >> 4) << 3);
  const int nb = n0 + (lane & 15);
#pragma unroll
  for (int nt = 0; nt < 4; ++nt)
#pragma unroll
    for (int r = 0; r < 8; ++r)
      Cb[(size_t)(mb + r) * ldc + nb + nt * 16] = alpha * acc[nt][r];
}

// ---- generic gemm, f16 store (optionally transposed), batched ----
template<bool NT, bool TR>
__global__ __launch_bounds__(128)
void k_gemm_f16o(const _Float16* A, int lda, long long sA,
                 const _Float16* B, int ldb, long long sB,
                 _Float16* C, int ldc, long long sC, int K, float alpha) {
  const _Float16* Ab = A + (size_t)blockIdx.z * sA;
  const _Float16* Bb = B + (size_t)blockIdx.z * sB;
  _Float16* Cb = C + (size_t)blockIdx.z * sC;
  const int wave = threadIdx.x >> 5;
  const int m0 = (blockIdx.y * 4 + wave) * 16;
  const int n0 = blockIdx.x * 64;
  v8f acc[4]; zero_acc(acc);
  gemm_tile<NT>(Ab, lda, Bb, ldb, K, m0, n0, acc);
  const int lane = threadIdx.x & 31;
  const int mb = m0 + ((lane >> 4) << 3);
  const int nb = n0 + (lane & 15);
#pragma unroll
  for (int nt = 0; nt < 4; ++nt)
#pragma unroll
    for (int r = 0; r < 8; ++r) {
      const int m = mb + r, n = nb + nt * 16;
      const _Float16 v = (_Float16)(alpha * acc[nt][r]);
      if (TR) Cb[(size_t)n * ldc + m] = v;
      else    Cb[(size_t)m * ldc + n] = v;
    }
}

// ---- QKV gemm: hn16[NP x 512] @ wqkvT[1536 x 512]^T, packed per-head output ----
__global__ __launch_bounds__(128)
void k_gemm_qkv(const _Float16* A, const _Float16* Bt,
                _Float16* q, _Float16* k, _Float16* v, _Float16* vT) {
  const int wave = threadIdx.x >> 5;
  const int m0 = (blockIdx.y * 4 + wave) * 16;
  const int n0 = blockIdx.x * 64;
  v8f acc[4]; zero_acc(acc);
  gemm_tile<true>(A, D, Bt, D, D, m0, n0, acc);
  const int lane = threadIdx.x & 31;
  const int mb = m0 + ((lane >> 4) << 3);
  const int nb = n0 + (lane & 15);
#pragma unroll
  for (int nt = 0; nt < 4; ++nt)
#pragma unroll
    for (int r = 0; r < 8; ++r) {
      const int m = mb + r, n = nb + nt * 16;
      const int which = n >> 9, rem = n & 511, hh = rem >> 6, d = rem & 63;
      const float val = acc[nt][r];
      const size_t off = ((size_t)hh * NP + m) * DH + d;
      if (which == 0)      q[off] = (_Float16)(val * 0.125f);   // dh^-0.5
      else if (which == 1) k[off] = (_Float16)val;
      else { v[off] = (_Float16)val; vT[((size_t)hh * DH + d) * NP + m] = (_Float16)val; }
    }
}

// ---- out-projection gemm: attn16[NP x 512] @ woutT^T + bias, add into h rows m-PADR ----
__global__ __launch_bounds__(128)
void k_gemm_outproj(const _Float16* A, const _Float16* Bt, const float* bias, float* h) {
  const int wave = threadIdx.x >> 5;
  const int m0 = (blockIdx.y * 4 + wave) * 16;
  const int n0 = blockIdx.x * 64;
  v8f acc[4]; zero_acc(acc);
  gemm_tile<true>(A, D, Bt, D, D, m0, n0, acc);
  const int lane = threadIdx.x & 31;
  const int mb = m0 + ((lane >> 4) << 3);
  const int nb = n0 + (lane & 15);
#pragma unroll
  for (int nt = 0; nt < 4; ++nt)
#pragma unroll
    for (int r = 0; r < 8; ++r) {
      const int m = mb + r, n = nb + nt * 16;
      if (m >= PADR)
        h[(size_t)(m - PADR) * D + n] += acc[nt][r] + bias[n];
    }
}

// =====================================================================
// elementwise / reduction kernels
// =====================================================================
__global__ void k_convert_T(const float* in, _Float16* out, int K, int N) {
  const int idx = blockIdx.x * blockDim.x + threadIdx.x;
  if (idx >= K * N) return;
  const int k = idx / N, n = idx % N;
  out[(size_t)n * K + k] = (_Float16)in[idx];
}

__global__ void k_assemble(const float* data, const float* w, const float* b,
                           const float* cls, float* h) {
  const long long idx = (long long)blockIdx.x * blockDim.x + threadIdx.x;
  if (idx >= (long long)N1 * D) return;
  const int r = (int)(idx / D), d = (int)(idx % D);
  if (r == 0) { h[idx] = cls[d]; return; }
  const int n = r - 1;
  float acc = b[d];
#pragma unroll
  for (int i = 0; i < 4; ++i) acc += data[(size_t)n * 4 + i] * w[i * D + d];
  h[idx] = fmaxf(acc, 0.f);
}

__global__ void k_layernorm(const float* h, const float* g, const float* bb, _Float16* out) {
  const int r = blockIdx.x;   // 0..NP-1
  const int t = threadIdx.x;  // 256
  __shared__ float red[256], red2[256];
  if (r < PADR) {
    out[(size_t)r * D + t]       = (_Float16)0.f;
    out[(size_t)r * D + t + 256] = (_Float16)0.f;
    return;
  }
  const float* x = h + (size_t)(r - PADR) * D;
  const float a0 = x[t], a1 = x[t + 256];
  red[t] = a0 + a1; red2[t] = a0 * a0 + a1 * a1;
  __syncthreads();
  for (int s = 128; s > 0; s >>= 1) { if (t < s) { red[t] += red[t + s]; red2[t] += red2[t + s]; } __syncthreads(); }
  const float mu = red[0] / D;
  const float var = red2[0] / D - mu * mu;
  const float rstd = rsqrtf(var + 1e-5f);
  out[(size_t)r * D + t]       = (_Float16)((a0 - mu) * rstd * g[t]       + bb[t]);
  out[(size_t)r * D + t + 256] = (_Float16)((a1 - mu) * rstd * g[t + 256] + bb[t + 256]);
}

__global__ void k_landmarks(const _Float16* q, const _Float16* k, _Float16* ql, _Float16* kl) {
  const int j  = blockIdx.x;   // landmark 0..255
  const int hh = blockIdx.y;   // head
  const int d  = threadIdx.x;  // 0..63
  const _Float16* qp = q + ((size_t)hh * NP + (size_t)j * LG) * DH + d;
  const _Float16* kp = k + ((size_t)hh * NP + (size_t)j * LG) * DH + d;
  float sq = 0.f, sk = 0.f;
  for (int i = 0; i < LG; ++i) { sq += (float)qp[(size_t)i * DH]; sk += (float)kp[(size_t)i * DH]; }
  ql[((size_t)hh * MM + j) * DH + d] = (_Float16)(sq / LG);
  kl[((size_t)hh * MM + j) * DH + d] = (_Float16)(sk / LG);
}

__global__ void k_softmax(const float* in, _Float16* out16, float* out32,
                          long long sBatch, int cols) {
  const int r = blockIdx.x, hh = blockIdx.y, t = threadIdx.x;
  const long long base = (long long)hh * sBatch + (long long)r * cols;
  const float* x = in + base;
  _Float16* o = out16 + base;
  __shared__ float red[256];
  float mx = -3.0e38f;
  for (int c = t; c < cols; c += 256) mx = fmaxf(mx, x[c]);
  red[t] = mx; __syncthreads();
  for (int s = 128; s > 0; s >>= 1) { if (t < s) red[t] = fmaxf(red[t], red[t + s]); __syncthreads(); }
  mx = red[0]; __syncthreads();
  float sum = 0.f;
  for (int c = t; c < cols; c += 256) sum += __expf(x[c] - mx);
  red[t] = sum; __syncthreads();
  for (int s = 128; s > 0; s >>= 1) { if (t < s) red[t] += red[t + s]; __syncthreads(); }
  const float inv = 1.f / red[0];
  for (int c = t; c < cols; c += 256) {
    const float v = __expf(x[c] - mx) * inv;
    o[c] = (_Float16)v;
    if (out32) out32[base + c] = v;
  }
}

__global__ void k_alpha(const float* x, float* ralpha) {  // x: [H][256][256]
  const int hh = blockIdx.x, t = threadIdx.x;
  const float* xb = x + (size_t)hh * MM * MM;
  float rs = 0.f, cs = 0.f;
  for (int j = 0; j < MM; ++j) { rs += fabsf(xb[(size_t)t * MM + j]); cs += fabsf(xb[(size_t)j * MM + t]); }
  __shared__ float r1[256], r2[256];
  r1[t] = rs; r2[t] = cs; __syncthreads();
  for (int s = 128; s > 0; s >>= 1) { if (t < s) { r1[t] = fmaxf(r1[t], r1[t + s]); r2[t] = fmaxf(r2[t], r2[t + s]); } __syncthreads(); }
  if (t == 0) ralpha[hh] = 1.f / (r1[0] * r2[0]);
}

__global__ void k_z0(const float* x, const float* ralpha, _Float16* z) {
  const long long idx = (long long)blockIdx.x * 256 + threadIdx.x;
  if (idx >= (long long)HH * MM * MM) return;
  const int hh = (int)(idx / (MM * MM)), rem = (int)(idx % (MM * MM));
  const int i = rem / MM, j = rem % MM;
  z[idx] = (_Float16)(x[(size_t)hh * MM * MM + (size_t)j * MM + i] * ralpha[hh]);
}

__global__ void k_identsub(const float* in, _Float16* out, float c) {
  const long long idx = (long long)blockIdx.x * 256 + threadIdx.x;
  if (idx >= (long long)HH * MM * MM) return;
  const int rem = (int)(idx % (MM * MM));
  const int i = rem / MM, j = rem % MM;
  out[idx] = (_Float16)((i == j ? c : 0.f) - in[idx]);
}

__global__ void k_cvt16(const float* in, _Float16* out, long long n, float alpha) {
  const long long idx = (long long)blockIdx.x * 256 + threadIdx.x;
  if (idx >= n) return;
  out[idx] = (_Float16)(alpha * in[idx]);
}

__global__ void k_resconv(float* attnC, const _Float16* v, const float* rw, const float* rb) {
  const long long idx = (long long)blockIdx.x * 256 + threadIdx.x;
  if (idx >= (long long)NP * D) return;
  const int r = (int)(idx / D), c = (int)(idx % D);
  const int hh = c >> 6, d = c & 63;
  const _Float16* vb = v + (size_t)hh * NP * DH + d;
  float s = rb[hh];
#pragma unroll
  for (int t = 0; t < 33; ++t) {
    const int rr = r + t - 16;
    if (rr >= 0 && rr < NP) s += (float)vb[(size_t)rr * DH] * rw[hh * 33 + t];
  }
  attnC[idx] += s;
}

__global__ void k_ppeg(const float* hA,
                       const float* w7, const float* b7,
                       const float* w5, const float* b5,
                       const float* w3, const float* b3, float* hB) {
  const long long idx = (long long)blockIdx.x * 256 + threadIdx.x;
  if (idx >= (long long)N1 * D) return;
  const int r = (int)(idx / D), c = (int)(idx % D);
  if (r == 0) { hB[idx] = hA[idx]; return; }
  const int p = r - 1, y = p / HS, x = p % HS;
  float acc = hA[idx] + b7[c] + b5[c] + b3[c];
  for (int ky = 0; ky < 7; ++ky) { const int yy = y + ky - 3; if (yy < 0 || yy >= HS) continue;
    for (int kx = 0; kx < 7; ++kx) { const int xx = x + kx - 3; if (xx < 0 || xx >= HS) continue;
      acc += hA[((size_t)(1 + yy * HS + xx)) * D + c] * w7[(size_t)c * 49 + ky * 7 + kx]; } }
  for (int ky = 0; ky < 5; ++ky) { const int yy = y + ky - 2; if (yy < 0 || yy >= HS) continue;
    for (int kx = 0; kx < 5; ++kx) { const int xx = x + kx - 2; if (xx < 0 || xx >= HS) continue;
      acc += hA[((size_t)(1 + yy * HS + xx)) * D + c] * w5[(size_t)c * 25 + ky * 5 + kx]; } }
  for (int ky = 0; ky < 3; ++ky) { const int yy = y + ky - 1; if (yy < 0 || yy >= HS) continue;
    for (int kx = 0; kx < 3; ++kx) { const int xx = x + kx - 1; if (xx < 0 || xx >= HS) continue;
      acc += hA[((size_t)(1 + yy * HS + xx)) * D + c] * w3[(size_t)c * 9 + ky * 3 + kx]; } }
  hB[idx] = acc;
}

__global__ void k_head(const float* h, const float* g, const float* bb,
                       const float* fw, const float* fb, float* out) {
  __shared__ float vsh[512];
  __shared__ float red[256];
  const int t = threadIdx.x;
  const float a0 = h[t], a1 = h[t + 256];
  red[t] = a0 + a1; __syncthreads();
  for (int s = 128; s > 0; s >>= 1) { if (t < s) red[t] += red[t + s]; __syncthreads(); }
  const float mu = red[0] / 512.f; __syncthreads();
  red[t] = (a0 - mu) * (a0 - mu) + (a1 - mu) * (a1 - mu); __syncthreads();
  for (int s = 128; s > 0; s >>= 1) { if (t < s) red[t] += red[t + s]; __syncthreads(); }
  const float rstd = rsqrtf(red[0] / 512.f + 1e-5f); __syncthreads();
  vsh[t]       = (a0 - mu) * rstd * g[t]       + bb[t];
  vsh[t + 256] = (a1 - mu) * rstd * g[t + 256] + bb[t + 256];
  __syncthreads();
  for (int c = 0; c < 4; ++c) {
    float s = 0.f;
    for (int d = t; d < 512; d += 256) s += vsh[d] * fw[d * 4 + c];
    red[t] = s; __syncthreads();
    for (int st = 128; st > 0; st >>= 1) { if (t < st) red[t] += red[t + st]; __syncthreads(); }
    if (t == 0) out[c] = red[0] + fb[c];
    __syncthreads();
  }
}

// =====================================================================
// host orchestration
// =====================================================================
struct WS {
  float *hA, *hB, *sc32, *attnC, *p32a, *p32b, *a2f32, *a3v32, *ralpha;
  _Float16 *hn16, *q16, *k16, *v16, *vT16, *ql16, *kl16, *sc16;
  _Float16 *x16, *xz16, *t16, *z[2], *a3v16, *WhT;
  _Float16 *wqkvT1, *woutT1, *wqkvT2, *woutT2;
};

static inline size_t al256(size_t x) { return (x + 255) & ~(size_t)255; }

static void attn_layer(float* h, const float* lng, const float* lnb,
                       const _Float16* qkvT, const _Float16* woutT, const float* outb,
                       const float* resw, const float* resb,
                       WS& w, hipStream_t st) {
  const long long sQ  = (long long)NP * DH;   // per-head q/k/v elems
  const long long sL  = (long long)MM * DH;   // per-head landmark elems
  const long long sM2 = (long long)MM * MM;
  const long long sS  = (long long)MM * NP;   // per-head a3 score elems
  const long long sA1 = (long long)NP * MM;   // per-head a1 score elems
  const int eb = ((int)((long long)HH * MM * MM) + 255) / 256;

  k_layernorm<<<NP, 256, 0, st>>>(h, lng, lnb, w.hn16);
  k_gemm_qkv<<<dim3(1536 / 64, NP / 64, 1), 128, 0, st>>>(w.hn16, qkvT, w.q16, w.k16, w.v16, w.vT16);
  k_landmarks<<<dim3(MM, HH), 64, 0, st>>>(w.q16, w.k16, w.ql16, w.kl16);

  // a2 = softmax(ql @ kl^T) ; x16 = f16 copy, a2f32 = f32 copy for pinv init
  k_gemm_f32<true><<<dim3(4, 4, HH), 128, 0, st>>>(w.ql16, DH, sL, w.kl16, DH, sL,
                                                   w.p32a, MM, sM2, DH, 1.f);
  k_softmax<<<dim3(MM, HH), 256, 0, st>>>(w.p32a, w.x16, w.a2f32, sM2, MM);

  // pinv init: z0 = x^T / (max|row|_1 * max|col|_1)
  k_alpha<<<HH, 256, 0, st>>>(w.a2f32, w.ralpha);
  k_z0<<<eb, 256, 0, st>>>(w.a2f32, w.ralpha, w.z[0]);

  // 6 Newton-Schulz iterations: z = 0.25 z (13I - xz(15I - xz(7I - xz)))
  int cur = 0;
  for (int it = 0; it < 6; ++it) {
    k_gemm_f32<false><<<dim3(4, 4, HH), 128, 0, st>>>(w.x16, MM, sM2, w.z[cur], MM, sM2,
                                                      w.p32a, MM, sM2, MM, 1.f);
    k_cvt16<<<eb, 256, 0, st>>>(w.p32a, w.xz16, (long long)HH * sM2, 1.f);
    k_identsub<<<eb, 256, 0, st>>>(w.p32a, w.t16, 7.f);
    k_gemm_f32<false><<<dim3(4, 4, HH), 128, 0, st>>>(w.xz16, MM, sM2, w.t16, MM, sM2,
                                                      w.p32b, MM, sM2, MM, 1.f);
    k_identsub<<<eb, 256, 0, st>>>(w.p32b, w.t16, 15.f);
    k_gemm_f32<false><<<dim3(4, 4, HH), 128, 0, st>>>(w.xz16, MM, sM2, w.t16, MM, sM2,
                                                      w.p32b, MM, sM2, MM, 1.f);
    k_identsub<<<eb, 256, 0, st>>>(w.p32b, w.t16, 13.f);
    k_gemm_f16o<false, false><<<dim3(4, 4, HH), 128, 0, st>>>(w.z[cur], MM, sM2, w.t16, MM, sM2,
                                                              w.z[1 - cur], MM, sM2, MM, 0.25f);
    cur ^= 1;
  }

  // a3 = softmax(ql @ k^T)  [H,256,NP];  a3v = a3 @ v  [H,256,64]
  k_gemm_f32<true><<<dim3(NP / 64, 4, HH), 128, 0, st>>>(w.ql16, DH, sL, w.k16, DH, sQ,
                                                         w.sc32, NP, sS, DH, 1.f);
  k_softmax<<<dim3(MM, HH), 256, 0, st>>>(w.sc32, w.sc16, nullptr, sS, NP);
  k_gemm_f32<true><<<dim3(1, 4, HH), 128, 0, st>>>(w.sc16, NP, sS, w.vT16, NP, (long long)DH * NP,
                                                   w.a3v32, DH, sL, NP, 1.f);
  k_cvt16<<<(int)(((long long)HH * sL + 255) / 256), 256, 0, st>>>(w.a3v32, w.a3v16,
                                                                   (long long)HH * sL, 1.f);
  // WhT = (Z @ a3v)^T  [H,64,256]
  k_gemm_f16o<false, true><<<dim3(1, 4, HH), 128, 0, st>>>(w.z[cur], MM, sM2, w.a3v16, DH, sL,
                                                           w.WhT, MM, (long long)DH * MM, MM, 1.f);
  // a1 = softmax(q @ kl^T)  [H,NP,256];  attn = a1 @ Wh -> attnC[:, head*64:...]
  k_gemm_f32<true><<<dim3(4, NP / 64, HH), 128, 0, st>>>(w.q16, DH, sQ, w.kl16, DH, sL,
                                                         w.sc32, MM, sA1, DH, 1.f);
  k_softmax<<<dim3(NP, HH), 256, 0, st>>>(w.sc32, w.sc16, nullptr, sA1, MM);
  k_gemm_f32<true><<<dim3(1, NP / 64, HH), 128, 0, st>>>(w.sc16, MM, sA1, w.WhT, MM,
                                                         (long long)DH * MM,
                                                         w.attnC, D, (long long)DH, MM, 1.f);
  // + depthwise 33x1 residual conv on v, + res_b
  const int gb = (int)(((long long)NP * D + 255) / 256);
  k_resconv<<<gb, 256, 0, st>>>(w.attnC, w.v16, resw, resb);
  // out-projection + bias, add into h (drops the PADR pad rows)
  k_cvt16<<<gb, 256, 0, st>>>(w.attnC, w.hn16, (long long)NP * D, 1.f);
  k_gemm_outproj<<<dim3(D / 64, NP / 64, 1), 128, 0, st>>>(w.hn16, woutT, outb, h);
}

extern "C" void kernel_launch(void* const* d_in, const int* in_sizes, int n_in,
                              void* d_out, int out_size, void* d_ws, size_t ws_size,
                              hipStream_t stream) {
  const float* data     = (const float*)d_in[0];
  const float* fc1_w    = (const float*)d_in[1];
  const float* fc1_b    = (const float*)d_in[2];
  const float* cls_tok  = (const float*)d_in[3];
  const float* l1_ln_g  = (const float*)d_in[4];
  const float* l1_ln_b  = (const float*)d_in[5];
  const float* l1_qkv   = (const float*)d_in[6];
  const float* l1_out_w = (const float*)d_in[7];
  const float* l1_out_b = (const float*)d_in[8];
  const float* l1_res_w = (const float*)d_in[9];
  const float* l1_res_b = (const float*)d_in[10];
  const float* ppeg_w7  = (const float*)d_in[11];
  const float* ppeg_b7  = (const float*)d_in[12];
  const float* ppeg_w5  = (const float*)d_in[13];
  const float* ppeg_b5  = (const float*)d_in[14];
  const float* ppeg_w3  = (const float*)d_in[15];
  const float* ppeg_b3  = (const float*)d_in[16];
  const float* l2_ln_g  = (const float*)d_in[17];
  const float* l2_ln_b  = (const float*)d_in[18];
  const float* l2_qkv   = (const float*)d_in[19];
  const float* l2_out_w = (const float*)d_in[20];
  const float* l2_out_b = (const float*)d_in[21];
  const float* l2_res_w = (const float*)d_in[22];
  const float* l2_res_b = (const float*)d_in[23];
  const float* norm_g   = (const float*)d_in[24];
  const float* norm_b   = (const float*)d_in[25];
  const float* fc2_w    = (const float*)d_in[26];
  const float* fc2_b    = (const float*)d_in[27];

  char* p = (char*)d_ws;
  auto grab = [&](size_t bytes) -> char* { char* r = p; p += al256(bytes); return r; };

  WS w;
  const size_t fH  = (size_t)N1 * D * sizeof(float);
  const size_t fS  = (size_t)HH * MM * NP * sizeof(float);
  const size_t fC  = (size_t)NP * D * sizeof(float);
  const size_t fM2 = (size_t)HH * MM * MM * sizeof(float);
  w.hA     = (float*)grab(fH);
  w.hB     = (float*)grab(fH);
  w.sc32   = (float*)grab(fS);
  w.attnC  = (float*)grab(fC);
  w.p32a   = (float*)grab(fM2);
  w.p32b   = (float*)grab(fM2);
  w.a2f32  = (float*)grab(fM2);
  w.a3v32  = (float*)grab((size_t)HH * MM * DH * sizeof(float));
  w.ralpha = (float*)grab(256);
  w.hn16   = (_Float16*)grab((size_t)NP * D * 2);
  w.q16    = (_Float16*)grab((size_t)HH * NP * DH * 2);
  w.k16    = (_Float16*)grab((size_t)HH * NP * DH * 2);
  w.v16    = (_Float16*)grab((size_t)HH * NP * DH * 2);
  w.vT16   = (_Float16*)grab((size_t)HH * NP * DH * 2);
  w.ql16   = (_Float16*)grab((size_t)HH * MM * DH * 2);
  w.kl16   = (_Float16*)grab((size_t)HH * MM * DH * 2);
  w.sc16   = (_Float16*)grab((size_t)HH * MM * NP * 2);
  w.x16    = (_Float16*)grab((size_t)HH * MM * MM * 2);
  w.xz16   = (_Float16*)grab((size_t)HH * MM * MM * 2);
  w.t16    = (_Float16*)grab((size_t)HH * MM * MM * 2);
  w.z[0]   = (_Float16*)grab((size_t)HH * MM * MM * 2);
  w.z[1]   = (_Float16*)grab((size_t)HH * MM * MM * 2);
  w.a3v16  = (_Float16*)grab((size_t)HH * MM * DH * 2);
  w.WhT    = (_Float16*)grab((size_t)HH * DH * MM * 2);
  w.wqkvT1 = (_Float16*)grab((size_t)1536 * D * 2);
  w.woutT1 = (_Float16*)grab((size_t)D * D * 2);
  w.wqkvT2 = (_Float16*)grab((size_t)1536 * D * 2);
  w.woutT2 = (_Float16*)grab((size_t)D * D * 2);

  // one-time weight transpose+convert to f16 (NT layout for contiguous B fragments)
  k_convert_T<<<(D * 1536 + 255) / 256, 256, 0, stream>>>(l1_qkv, w.wqkvT1, D, 1536);
  k_convert_T<<<(D * D + 255) / 256, 256, 0, stream>>>(l1_out_w, w.woutT1, D, D);
  k_convert_T<<<(D * 1536 + 255) / 256, 256, 0, stream>>>(l2_qkv, w.wqkvT2, D, 1536);
  k_convert_T<<<(D * D + 255) / 256, 256, 0, stream>>>(l2_out_w, w.woutT2, D, D);

  // fc1 + relu + cls token
  k_assemble<<<(int)(((long long)N1 * D + 255) / 256), 256, 0, stream>>>(data, fc1_w, fc1_b,
                                                                         cls_tok, w.hA);
  // layer 1 (residual add into hA)
  attn_layer(w.hA, l1_ln_g, l1_ln_b, w.wqkvT1, w.woutT1, l1_out_b, l1_res_w, l1_res_b, w, stream);
  // PPEG: hA -> hB
  k_ppeg<<<(int)(((long long)N1 * D + 255) / 256), 256, 0, stream>>>(w.hA, ppeg_w7, ppeg_b7,
                                                                     ppeg_w5, ppeg_b5,
                                                                     ppeg_w3, ppeg_b3, w.hB);
  // layer 2 (residual add into hB)
  attn_layer(w.hB, l2_ln_g, l2_ln_b, w.wqkvT2, w.woutT2, l2_out_b, l2_res_w, l2_res_b, w, stream);
  // final LN on cls + fc2
  k_head<<<1, 256, 0, stream>>>(w.hB, norm_g, norm_b, fc2_w, fc2_b, (float*)d_out);
  (void)in_sizes; (void)n_in; (void)out_size; (void)ws_size;
}
